// ECForGraphTCN_13778255085990
// MI455X (gfx1250) — matrix operand order repaired and checked by reading, not compiled, not run on hardware
//
#include <hip/hip_runtime.h>
#include <hip/hip_bf16.h>
#include <math.h>

// ---------------------------------------------------------------------------
// ECForGraphTCN on MI455X (gfx1250, wave32, WMMA).
// fp16 h/e storage, v_wmma_f32_16x16x32_f16 for every MLP layer, f32 atomics
// for segment_sum into L2-resident agg. All global<->LDS movement vectorized
// as b128; contiguous e-tiles use GLOBAL_LOAD_ASYNC_TO_LDS_B128 (ASYNCcnt).
// 4 waves / 128-thread block, one 16-row tile per wave.
// ---------------------------------------------------------------------------

#define N_NODES 200000
#define N_EDGES 2000000
#define NDIM    32
#define EDIM    32
#define HID     64
#define L_EC    6

typedef __attribute__((ext_vector_type(16))) _Float16 v16h;
typedef __attribute__((ext_vector_type(4)))  _Float16 v4h;
typedef __attribute__((ext_vector_type(8)))  float    v8f;

// ---------------- WMMA fragment helpers (CDNA5 documented layouts) ----------

// A fragment: 16x32 f16 tile, row-major in LDS, row stride ld (halfs).
// Lane L<16 : row M=L,    K = {0..7, 16..23};  Lane L>=16: K = {8..15, 24..31}
__device__ __forceinline__ v16h wmma_load_a(const _Float16* s, int ld) {
    const int lane = threadIdx.x & 31;
    const int row  = lane & 15;
    const int kh   = (lane >> 4) << 3;
    v16h a;
#pragma unroll
    for (int p = 0; p < 8; ++p) {
        const int kb = ((p < 4) ? (2 * p) : (8 + 2 * p)) + kh;
        a[2 * p]     = s[row * ld + kb];
        a[2 * p + 1] = s[row * ld + kb + 1];
    }
    return a;
}

// B fragment: 32x16 f16 weight tile W[K][N], row-major, row stride ld.
// Lane L<16 : col N=L, K = {0..7, 16..23};  Lane L>=16: N=L-16, K = {8..15, 24..31}
__device__ __forceinline__ v16h wmma_load_b(const _Float16* w, int ld) {
    const int lane = threadIdx.x & 31;
    const int col  = lane & 15;
    const int kh   = (lane >> 4) << 3;
    v16h b;
#pragma unroll
    for (int p = 0; p < 8; ++p) {
        const int kb = ((p < 4) ? (2 * p) : (8 + 2 * p)) + kh;
        b[2 * p]     = w[kb * ld + col];
        b[2 * p + 1] = w[(kb + 1) * ld + col];
    }
    return b;
}

// C/D: 16x16 f32; lane -> N = lane&15, VGPR v -> M = v + 8*(lane>>4)
__device__ __forceinline__ void wmma_store_d(v8f c, _Float16* out, int ld, bool relu) {
    const int lane = threadIdx.x & 31;
    const int n    = lane & 15;
    const int mh   = (lane >> 4) << 3;
#pragma unroll
    for (int v = 0; v < 8; ++v) {
        float x = c[v];
        if (relu) x = fmaxf(x, 0.f);
        out[(v + mh) * ld + n] = (_Float16)x;
    }
}

__device__ __forceinline__ v8f wmma32(v16h a, v16h b, v8f c) {
    return __builtin_amdgcn_wmma_f32_16x16x32_f16(false, a, false, b, (short)0, c,
                                                  false, false);
}

// OUT[16][NO] = act(IN[16][KI] @ W[KI][NO] + b)
template <int KI, int NO, bool RELU>
__device__ __forceinline__ void tile_gemm(const _Float16* in, int ldin,
                                          const _Float16* W, int ldw,
                                          const float* bias,
                                          _Float16* out, int ldout) {
    const int lane = threadIdx.x & 31;
#pragma unroll
    for (int n = 0; n < NO / 16; ++n) {
        const float bv = bias ? bias[n * 16 + (lane & 15)] : 0.f;
        v8f c = {bv, bv, bv, bv, bv, bv, bv, bv};
#pragma unroll
        for (int k = 0; k < KI / 32; ++k) {
            v16h a = wmma_load_a(in + k * 32, ldin);
            v16h b = wmma_load_b(W + (k * 32) * ldw + n * 16, ldw);
            c = wmma32(a, b, c);
        }
        wmma_store_d(c, out + n * 16, ldout, RELU);
    }
}

// ---------------- async global->LDS (GLOBAL_LOAD_ASYNC_TO_LDS_B128) ---------
// Low 32 bits of a generic shared-memory address are the wave-relative LDS
// byte offset (ISA 10.2 aperture mapping); VDST supplies it per-lane.
__device__ __forceinline__ void async_copy16(void* lds, const void* gsrc) {
    const unsigned ldsOff = (unsigned)(unsigned long long)lds;
    asm volatile("global_load_async_to_lds_b128 %0, %1, off"
                 :: "v"(ldsOff), "v"(gsrc) : "memory");
}
__device__ __forceinline__ void async_wait0() {
#if __has_builtin(__builtin_amdgcn_s_wait_asynccnt)
    __builtin_amdgcn_s_wait_asynccnt(0);
#else
    asm volatile("s_wait_asynccnt 0x0" ::: "memory");
#endif
}

// ---------------- weight staging: f32 global -> f16 LDS, vectorized ---------
__device__ __forceinline__ void stage_w(const float* __restrict__ g,
                                        _Float16* s, int nElem) {
    const float4* g4 = (const float4*)g;
    for (int i = threadIdx.x; i < nElem / 4; i += 128) {
        const float4 v = g4[i];
        ((v4h*)s)[i] = (v4h){(_Float16)v.x, (_Float16)v.y,
                             (_Float16)v.z, (_Float16)v.w};
    }
}

// ---------------------------- utility ---------------------------------------
__global__ void zero_f32_kernel(float* __restrict__ p, long long n) {
    long long i = (long long)blockIdx.x * blockDim.x + threadIdx.x;
    const long long stride = (long long)gridDim.x * blockDim.x;
    for (; i < n; i += stride) p[i] = 0.f;
}

// ------------------------- encoders -----------------------------------------
__global__ __launch_bounds__(128) void node_enc_kernel(
    const float* __restrict__ x, const float* __restrict__ W0,
    const float* __restrict__ W1, _Float16* __restrict__ h) {
    __shared__ __align__(16) _Float16 sW0[32 * 64];   // K padded 14 -> 32
    __shared__ __align__(16) _Float16 sW1[64 * 32];
    __shared__ __align__(16) _Float16 sA0[4][16 * 32];
    __shared__ __align__(16) _Float16 sA1[4][16 * 64];

    for (int i = threadIdx.x; i < 32 * 64; i += 128) {
        const int k = i >> 6;
        sW0[i] = (k < 14) ? (_Float16)W0[k * 64 + (i & 63)] : (_Float16)0.f;
    }
    stage_w(W1, sW1, 64 * 32);
    __syncthreads();

    const int w = threadIdx.x >> 5, lane = threadIdx.x & 31;
    const int tile = (blockIdx.x * 4 + w) * 16;
    if (tile >= N_NODES) return;                      // wave-uniform
    _Float16* A0 = sA0[w];
    _Float16* A1 = sA1[w];

    for (int idx = lane; idx < 16 * 32; idx += 32) {  // x rows are 14 f32 (unaligned)
        const int r = idx >> 5, c = idx & 31;
        const int rr = min(tile + r, N_NODES - 1);
        A0[idx] = (c < 14) ? (_Float16)x[rr * 14 + c] : (_Float16)0.f;
    }
    tile_gemm<32, 64, true>(A0, 32, sW0, 64, nullptr, A1, 64);
    tile_gemm<64, 32, true>(A1, 64, sW1, 32, nullptr, A0, 32);
#pragma unroll
    for (int t = lane; t < 64; t += 32) {
        const int r = t >> 2, q = t & 3;
        const int rr = tile + r;
        if (rr < N_NODES)
            *((int4*)(h + (size_t)rr * NDIM) + q) = *(const int4*)(A0 + r * 32 + 8 * q);
    }
}

__global__ __launch_bounds__(128) void edge_enc_kernel(
    const float* __restrict__ ea, const float* __restrict__ W0,
    const float* __restrict__ W1, _Float16* __restrict__ e) {
    __shared__ __align__(16) _Float16 sW0[32 * 64];   // K padded 4 -> 32
    __shared__ __align__(16) _Float16 sW1[64 * 32];
    __shared__ __align__(16) _Float16 sA0[4][16 * 32];
    __shared__ __align__(16) _Float16 sA1[4][16 * 64];

    for (int i = threadIdx.x; i < 32 * 64; i += 128) {
        const int k = i >> 6;
        sW0[i] = (k < 4) ? (_Float16)W0[k * 64 + (i & 63)] : (_Float16)0.f;
    }
    stage_w(W1, sW1, 64 * 32);
    __syncthreads();

    const int w = threadIdx.x >> 5, lane = threadIdx.x & 31;
    const int tile = (blockIdx.x * 4 + w) * 16;
    if (tile >= N_EDGES) return;
    _Float16* A0 = sA0[w];
    _Float16* A1 = sA1[w];

    if (lane < 16) {                                  // one row per lane, 4 f32 = 16B
        const int rr = min(tile + lane, N_EDGES - 1);
        const float4 v = *(const float4*)(ea + (size_t)rr * 4);
        *(v4h*)(A0 + lane * 32) = (v4h){(_Float16)v.x, (_Float16)v.y,
                                        (_Float16)v.z, (_Float16)v.w};
        *(v4h*)(A0 + lane * 32 + 4) = (v4h){(_Float16)0.f, (_Float16)0.f,
                                            (_Float16)0.f, (_Float16)0.f};
        *((int4*)(A0 + lane * 32 + 8))  = make_int4(0, 0, 0, 0);
        *((int4*)(A0 + lane * 32 + 16)) = make_int4(0, 0, 0, 0);
        *((int4*)(A0 + lane * 32 + 24)) = make_int4(0, 0, 0, 0);
    }
    tile_gemm<32, 64, true>(A0, 32, sW0, 64, nullptr, A1, 64);
    tile_gemm<64, 32, true>(A1, 64, sW1, 32, nullptr, A0, 32);
#pragma unroll
    for (int t = lane; t < 64; t += 32) {
        const int r = t >> 2, q = t & 3;
        const int rr = tile + r;
        if (rr < N_EDGES)
            *((int4*)(e + (size_t)rr * EDIM) + q) = *(const int4*)(A0 + r * 32 + 8 * q);
    }
}

// ------------------------- relational (edge) kernel --------------------------
__global__ __launch_bounds__(128) void rel_kernel(
    const _Float16* __restrict__ h, _Float16* __restrict__ e,
    float* __restrict__ agg, const int* __restrict__ ei,
    const float* __restrict__ W0, const float* __restrict__ b0,
    const float* __restrict__ W1, const float* __restrict__ b1,
    const float* __restrict__ W2, const float* __restrict__ b2) {
    __shared__ __align__(16) _Float16 sW0[96 * 64];
    __shared__ __align__(16) _Float16 sW1[64 * 64];
    __shared__ __align__(16) _Float16 sW2[64 * 32];
    __shared__ float sB0[64], sB1[64], sB2[32];
    __shared__ __align__(16) _Float16 sM[4][16 * 96];
    __shared__ __align__(16) _Float16 sT1[4][16 * 64];
    __shared__ __align__(16) _Float16 sT2[4][16 * 64];
    __shared__ int sDst[4][16], sSrc[4][16];

    stage_w(W0, sW0, 96 * 64);
    stage_w(W1, sW1, 64 * 64);
    stage_w(W2, sW2, 64 * 32);
    for (int i = threadIdx.x; i < 64; i += 128) { sB0[i] = b0[i]; sB1[i] = b1[i]; }
    for (int i = threadIdx.x; i < 32; i += 128) sB2[i] = b2[i];
    __syncthreads();

    const int w = threadIdx.x >> 5, lane = threadIdx.x & 31;
    const int tile = (blockIdx.x * 4 + w) * 16;
    if (tile >= N_EDGES) return;
    _Float16* M  = sM[w];
    _Float16* T1 = sT1[w];
    _Float16* T2 = sT2[w];

    if (lane < 16) {
        const int ec = min(tile + lane, N_EDGES - 1);
        sDst[w][lane] = ei[N_EDGES + ec];             // dst
        sSrc[w][lane] = ei[ec];                       // src
    }

    // e cols [64,96): contiguous stream -> async global->LDS (ASYNCcnt)
#pragma unroll
    for (int t = lane; t < 64; t += 32) {
        const int r = t >> 2, q = t & 3;
        const int ec = min(tile + r, N_EDGES - 1);
        async_copy16(M + r * 96 + 64 + 8 * q, e + (size_t)ec * EDIM + 8 * q);
    }
    // h gathers: rows are 64B; move as b128 chunks (L2-resident table)
#pragma unroll
    for (int t = lane; t < 64; t += 32) {
        const int r = t >> 2, q = t & 3;
        *((int4*)(M + r * 96) + 0 * 0 + q) = 0 ? make_int4(0,0,0,0)
            : *((const int4*)(h + (size_t)sDst[w][r] * NDIM) + q);        // x_i
        *(int4*)(M + r * 96 + 32 + 8 * q) =
              *((const int4*)(h + (size_t)sSrc[w][r] * NDIM) + q);        // x_j
    }
    async_wait0();

    tile_gemm<96, 64, true>(M, 96, sW0, 64, sB0, T1, 64);
    tile_gemm<64, 64, true>(T1, 64, sW1, 64, sB1, T2, 64);

    // final linear 64 -> 32; scatter-add e_new, residual into M's e region
    const int ncol = lane & 15;
    const int mh   = (lane >> 4) << 3;
#pragma unroll
    for (int n = 0; n < 2; ++n) {
        const float bv = sB2[n * 16 + ncol];
        v8f c = {bv, bv, bv, bv, bv, bv, bv, bv};
#pragma unroll
        for (int k = 0; k < 2; ++k)
            c = wmma32(wmma_load_a(T2 + k * 32, 64),
                       wmma_load_b(sW2 + (k * 32) * 32 + n * 16, 32), c);
        const int col = n * 16 + ncol;
#pragma unroll
        for (int v = 0; v < 8; ++v) {
            const int r = v + mh;
            if (tile + r < N_EDGES) {
                const float en = c[v];
                atomicAdd(&agg[(size_t)sDst[w][r] * EDIM + col], en);
                const float eo = (float)M[r * 96 + 64 + col];
                M[r * 96 + 64 + col] = (_Float16)(0.5f * eo + 0.5f * fmaxf(en, 0.f));
            }
        }
    }
    // vectorized writeback of updated e rows
#pragma unroll
    for (int t = lane; t < 64; t += 32) {
        const int r = t >> 2, q = t & 3;
        const int eidx = tile + r;
        if (eidx < N_EDGES)
            *((int4*)(e + (size_t)eidx * EDIM) + q) =
                *(const int4*)(M + r * 96 + 64 + 8 * q);
    }
}

// ------------------------- object (node) kernel ------------------------------
__global__ __launch_bounds__(128) void obj_kernel(
    _Float16* __restrict__ h, const float* __restrict__ agg,
    const float* __restrict__ W0, const float* __restrict__ b0,
    const float* __restrict__ W1, const float* __restrict__ b1,
    const float* __restrict__ W2, const float* __restrict__ b2) {
    __shared__ __align__(16) _Float16 sW0[64 * 64];
    __shared__ __align__(16) _Float16 sW1[64 * 64];
    __shared__ __align__(16) _Float16 sW2[64 * 32];
    __shared__ float sB0[64], sB1[64], sB2[32];
    __shared__ __align__(16) _Float16 sT0[4][16 * 64];
    __shared__ __align__(16) _Float16 sT1[4][16 * 64];
    __shared__ __align__(16) _Float16 sT2[4][16 * 64];

    stage_w(W0, sW0, 64 * 64);
    stage_w(W1, sW1, 64 * 64);
    stage_w(W2, sW2, 64 * 32);
    for (int i = threadIdx.x; i < 64; i += 128) { sB0[i] = b0[i]; sB1[i] = b1[i]; }
    for (int i = threadIdx.x; i < 32; i += 128) sB2[i] = b2[i];
    __syncthreads();

    const int w = threadIdx.x >> 5, lane = threadIdx.x & 31;
    const int tile = (blockIdx.x * 4 + w) * 16;
    if (tile >= N_NODES) return;
    _Float16* T0 = sT0[w];
    _Float16* T1 = sT1[w];
    _Float16* T2 = sT2[w];

    // cols [0,32) = h (b128 copies); cols [32,64) = (f16)agg (float4 -> 4 halfs)
#pragma unroll
    for (int t = lane; t < 64; t += 32) {
        const int r = t >> 2, q = t & 3;
        const int rr = min(tile + r, N_NODES - 1);
        *(int4*)(T0 + r * 64 + 8 * q) = *((const int4*)(h + (size_t)rr * NDIM) + q);
    }
#pragma unroll
    for (int t = lane; t < 128; t += 32) {
        const int r = t >> 3, q = t & 7;
        const int rr = min(tile + r, N_NODES - 1);
        const float4 v = *((const float4*)(agg + (size_t)rr * EDIM) + q);
        *(v4h*)(T0 + r * 64 + 32 + 4 * q) = (v4h){(_Float16)v.x, (_Float16)v.y,
                                                  (_Float16)v.z, (_Float16)v.w};
    }

    tile_gemm<64, 64, true>(T0, 64, sW0, 64, sB0, T1, 64);
    tile_gemm<64, 64, true>(T1, 64, sW1, 64, sB1, T2, 64);

    const int ncol = lane & 15;
    const int mh   = (lane >> 4) << 3;
#pragma unroll
    for (int n = 0; n < 2; ++n) {
        const float bv = sB2[n * 16 + ncol];
        v8f c = {bv, bv, bv, bv, bv, bv, bv, bv};
#pragma unroll
        for (int k = 0; k < 2; ++k)
            c = wmma32(wmma_load_a(T2 + k * 32, 64),
                       wmma_load_b(sW2 + (k * 32) * 32 + n * 16, 32), c);
        const int col = n * 16 + ncol;                // in [0,32)
#pragma unroll
        for (int v = 0; v < 8; ++v) {
            const int r = v + mh;
            if (tile + r < N_NODES) {
                const float hn = c[v];
                const float ho = (float)T0[r * 64 + col];
                T0[r * 64 + col] = (_Float16)(0.5f * ho + 0.5f * fmaxf(hn, 0.f));
            }
        }
    }
#pragma unroll
    for (int t = lane; t < 64; t += 32) {
        const int r = t >> 2, q = t & 3;
        const int rr = tile + r;
        if (rr < N_NODES)
            *((int4*)(h + (size_t)rr * NDIM) + q) = *(const int4*)(T0 + r * 64 + 8 * q);
    }
}

// ----------------------------- head kernel -----------------------------------
__global__ __launch_bounds__(128) void head_kernel(
    const _Float16* __restrict__ e,
    const float* __restrict__ W0, const float* __restrict__ b0,
    const float* __restrict__ W1, const float* __restrict__ b1,
    const float* __restrict__ W2, const float* __restrict__ b2,
    float* __restrict__ out) {
    __shared__ __align__(16) _Float16 sW0[32 * 64];
    __shared__ __align__(16) _Float16 sW1[64 * 64];
    __shared__ float sB0[64], sB1[64], sW2f[64];
    __shared__ float sB2v;
    __shared__ __align__(16) _Float16 sT0[4][16 * 32];
    __shared__ __align__(16) _Float16 sT1[4][16 * 64];
    __shared__ __align__(16) _Float16 sT2[4][16 * 64];

    stage_w(W0, sW0, 32 * 64);
    stage_w(W1, sW1, 64 * 64);
    for (int i = threadIdx.x; i < 64; i += 128) { sB0[i] = b0[i]; sB1[i] = b1[i]; sW2f[i] = W2[i]; }
    if (threadIdx.x == 0) sB2v = b2[0];
    __syncthreads();

    const int w = threadIdx.x >> 5, lane = threadIdx.x & 31;
    const int tile = (blockIdx.x * 4 + w) * 16;
    if (tile >= N_EDGES) return;
    _Float16* T0 = sT0[w];
    _Float16* T1 = sT1[w];
    _Float16* T2 = sT2[w];

#pragma unroll
    for (int t = lane; t < 64; t += 32) {
        const int r = t >> 2, q = t & 3;
        const int rr = min(tile + r, N_EDGES - 1);
        async_copy16(T0 + r * 32 + 8 * q, e + (size_t)rr * EDIM + 8 * q);
    }
    async_wait0();

    tile_gemm<32, 64, true>(T0, 32, sW0, 64, sB0, T1, 64);
    tile_gemm<64, 64, true>(T1, 64, sW1, 64, sB1, T2, 64);

    if (lane < 16) {
        const int eidx = tile + lane;
        if (eidx < N_EDGES) {
            float acc = sB2v;
#pragma unroll
            for (int k = 0; k < 64; ++k) acc += (float)T2[lane * 64 + k] * sW2f[k];
            out[eidx] = 1.f / (1.f + expf(-acc));
        }
    }
}

// ----------------------------- launcher --------------------------------------
// d_in (setup_inputs insertion order; params pytree in insertion order):
//  0:x  1:edge_attr  2,3:node_enc W0,W1  4,5:edge_enc W0,W1
//  6+12l+{0..11}: relW0,relB0,relW1,relB1,relW2,relB2,objW0,objB0,objW1,objB1,objW2,objB2
//  78..83: head W0,b0,W1,b1,W2,b2      84: edge_index [2,E] int32
extern "C" void kernel_launch(void* const* d_in, const int* in_sizes, int n_in,
                              void* d_out, int out_size, void* d_ws, size_t ws_size,
                              hipStream_t stream) {
    (void)in_sizes; (void)n_in; (void)out_size; (void)ws_size;

    const float* x   = (const float*)d_in[0];
    const float* ea  = (const float*)d_in[1];
    const int*   ei  = (const int*)d_in[84];
    float*       out = (float*)d_out;

    char* ws = (char*)d_ws;
    _Float16* h   = (_Float16*)ws;                                 // 12.8 MB (L2-resident)
    float*    agg = (float*)(ws + (size_t)N_NODES * NDIM * 2);     // 25.6 MB (L2-resident)
    _Float16* e   = (_Float16*)(ws + (size_t)N_NODES * NDIM * 2
                                   + (size_t)N_NODES * EDIM * 4);  // 128 MB  (HBM stream)

    const int nodeBlocks = (N_NODES / 16 + 3) / 4;   // 3125
    const int edgeBlocks = (N_EDGES / 16 + 3) / 4;   // 31250

    node_enc_kernel<<<nodeBlocks, 128, 0, stream>>>(
        x, (const float*)d_in[2], (const float*)d_in[3], h);
    edge_enc_kernel<<<edgeBlocks, 128, 0, stream>>>(
        ea, (const float*)d_in[4], (const float*)d_in[5], e);

    for (int l = 0; l < L_EC; ++l) {
        const int p = 6 + 12 * l;
        zero_f32_kernel<<<2048, 256, 0, stream>>>(agg, (long long)N_NODES * EDIM);
        rel_kernel<<<edgeBlocks, 128, 0, stream>>>(
            h, e, agg, ei,
            (const float*)d_in[p + 0], (const float*)d_in[p + 1],
            (const float*)d_in[p + 2], (const float*)d_in[p + 3],
            (const float*)d_in[p + 4], (const float*)d_in[p + 5]);
        obj_kernel<<<nodeBlocks, 128, 0, stream>>>(
            h, agg,
            (const float*)d_in[p + 6], (const float*)d_in[p + 7],
            (const float*)d_in[p + 8], (const float*)d_in[p + 9],
            (const float*)d_in[p + 10], (const float*)d_in[p + 11]);
    }

    head_kernel<<<edgeBlocks, 128, 0, stream>>>(
        e,
        (const float*)d_in[78], (const float*)d_in[79],
        (const float*)d_in[80], (const float*)d_in[81],
        (const float*)d_in[82], (const float*)d_in[83],
        out);
}
